// CRNN_25039659336087
// MI455X (gfx1250) — compile-verified, matrix operand order checked by reading, not run on hardware
//
#include <hip/hip_runtime.h>
#include <hip/hip_bf16.h>

// ---------- problem constants ----------
#define BB   32
#define TT   1024
#define MM   40
#define CD   128
#define HH   256
#define CC   10
#define G3H  768          // 3*H
#define KCONV 1152        // CD*9
#define TP   (TT + 2)     // padded time

typedef _Float16 h16_t;
typedef __attribute__((ext_vector_type(16))) _Float16 v16h;
typedef __attribute__((ext_vector_type(8)))  _Float16 v8h;
typedef __attribute__((ext_vector_type(8)))  float    v8f;

static __device__ __forceinline__ v8f zero8f() {
    v8f r;
#pragma unroll
    for (int i = 0; i < 8; ++i) r[i] = 0.0f;
    return r;
}
static __device__ __forceinline__ v8h ld8h(const h16_t* p) {
    return *reinterpret_cast<const v8h*>(p);
}
static __device__ __forceinline__ v16h ld16h(const h16_t* p) {
    return *reinterpret_cast<const v16h*>(p);
}
static __device__ __forceinline__ v16h combine16(v8h lo, v8h hi) {
    v16h r;
#pragma unroll
    for (int i = 0; i < 8; ++i) { r[i] = lo[i]; r[i + 8] = hi[i]; }
    return r;
}
static __device__ __forceinline__ v8f wmma_f16(v16h a, v16h b, v8f c) {
    return __builtin_amdgcn_wmma_f32_16x16x32_f16(
        false, a, false, b, (short)0, c, false, false);
}
static __device__ __forceinline__ float sigmoidf_(float x) {
    return 1.0f / (1.0f + __expf(-x));
}

// ---------- workspace layout (bytes) ----------
#define OFF_F1P   ((size_t)0)            // [B][TP][10][CD] f16 : 84,049,920
#define OFF_F2P   ((size_t)84049920)     // [B][TP][4][CD]  f16 : 33,619,968
#define OFF_FEAT  ((size_t)117669888)    // [T*B][CD] f16       : 8,388,608
#define OFF_GIX   ((size_t)126058496)    // [T*B][768] f32      : 100,663,296
#define OFF_W2F   ((size_t)226721792)    // frag-major          : 294,912
#define OFF_W3F   ((size_t)227016704)    // frag-major          : 294,912
#define OFF_WIHXF ((size_t)227311616)    // [48][4][32][16] f16 : 196,608
#define OFF_WHHF  ((size_t)227508224)    // [48][8][32][16] f16 : 393,216

// =====================================================================
// Zero fill (v8h stores), n8 = count of 8-half chunks
// =====================================================================
__global__ void zero_f16(h16_t* __restrict__ p, size_t n8) {
    size_t id = (size_t)blockIdx.x * blockDim.x + threadIdx.x;
    if (id >= n8) return;
    v8h z;
#pragma unroll
    for (int i = 0; i < 8; ++i) z[i] = (_Float16)0.0f;
    *reinterpret_cast<v8h*>(p + id * 8) = z;
}

// =====================================================================
// Fragment-major weight packing.
//   n = nt*16 + (lane&15),  k = kb*32 + (lane>>4)*8 + (i&7) + ((i>>3)<<4)
// =====================================================================
__global__ void pack_conv_frag(const float* __restrict__ W, h16_t* __restrict__ dst) {
    int id = blockIdx.x * blockDim.x + threadIdx.x;
    if (id >= 8 * 36 * 32 * 16) return;
    int i = id & 15, lane = (id >> 4) & 31;
    int fk = id >> 9;                   // nt*36 + kb
    int kb = fk % 36, nt = fk / 36;
    int k = kb * 32 + ((lane >> 4) << 3) + (i & 7) + ((i >> 3) << 4);
    int n = nt * 16 + (lane & 15);
    int ci = k & 127, tap = k >> 7;
    dst[id] = (h16_t)W[(size_t)n * KCONV + ci * 9 + tap];
}
__global__ void pack_wihx_frag(const float* __restrict__ Wih, h16_t* __restrict__ dst) {
    int id = blockIdx.x * blockDim.x + threadIdx.x;
    if (id >= 48 * 4 * 32 * 16) return;
    int i = id & 15, lane = (id >> 4) & 31;
    int fk = id >> 9;
    int kb = fk & 3, nt = fk >> 2;
    int k = kb * 32 + ((lane >> 4) << 3) + (i & 7) + ((i >> 3) << 4);
    int g = nt * 16 + (lane & 15);
    dst[id] = (h16_t)Wih[(size_t)g * (CD + CC) + k];
}
__global__ void pack_whh_frag(const float* __restrict__ Whh, h16_t* __restrict__ dst) {
    int id = blockIdx.x * blockDim.x + threadIdx.x;
    if (id >= 48 * 8 * 32 * 16) return;
    int i = id & 15, lane = (id >> 4) & 31;
    int fk = id >> 9;
    int kb = fk & 7, nt = fk >> 3;
    int k = kb * 32 + ((lane >> 4) << 3) + (i & 7) + ((i >> 3) << 4);
    int g = nt * 16 + (lane & 15);
    dst[id] = (h16_t)Whh[(size_t)g * HH + k];
}

// =====================================================================
// Conv1: direct VALU, fused ReLU + pool5 (mel 40->8), writes padded f1p
// =====================================================================
__global__ __launch_bounds__(128) void conv1_kernel(
    const float* __restrict__ x, const float* __restrict__ W1,
    const float* __restrict__ b1, h16_t* __restrict__ f1p) {
    int bid = blockIdx.x;
    int b = bid >> 10, t = bid & 1023;
    __shared__ float sx[3][MM];
    int tid = threadIdx.x;
    for (int i = tid; i < 3 * MM; i += 128) {
        int r = i / MM, mm = i % MM;
        int tt = t + r - 1;
        sx[r][mm] = (tt >= 0 && tt < TT) ? x[((size_t)b * TT + tt) * MM + mm] : 0.0f;
    }
    __syncthreads();
    int co = tid;
    float w[9];
#pragma unroll
    for (int i = 0; i < 9; ++i) w[i] = W1[co * 9 + i];
    float bias = b1[co];
    for (int mp = 0; mp < 8; ++mp) {
        float mx = -3.4e38f;
        for (int s = 0; s < 5; ++s) {
            int mm = mp * 5 + s;
            float acc = 0.0f;
#pragma unroll
            for (int dt = 0; dt < 3; ++dt)
#pragma unroll
                for (int dm = 0; dm < 3; ++dm) {
                    int m2 = mm + dm - 1;
                    if (m2 >= 0 && m2 < MM) acc += sx[dt][m2] * w[dt * 3 + dm];
                }
            mx = fmaxf(mx, acc + bias);
        }
        f1p[(((size_t)b * TP + (t + 1)) * 10 + (mp + 1)) * CD + co] = (h16_t)fmaxf(mx, 0.0f);
    }
}

// =====================================================================
// Conv2 implicit GEMM: one wave = one 16-position M-tile x ALL 8 N-tiles.
// Batched B-fragment loads per k-block for load/WMMA overlap.
// =====================================================================
__global__ __launch_bounds__(256) void conv2_kernel(
    const h16_t* __restrict__ f1p, const h16_t* __restrict__ w2f,
    const float* __restrict__ b2, h16_t* __restrict__ f2p) {
    int mt = (blockIdx.x * blockDim.x + threadIdx.x) >> 5;   // 0..16383
    int lane = threadIdx.x & 31;
    int lm = lane & 15, hi = lane >> 4;
    int p = mt * 16 + lm;
    int b = p >> 13, t = (p >> 3) & 1023, m = p & 7;
    const h16_t* abase = f1p + (((size_t)b * TP + t) * 10 + m) * CD + hi * 8;
    const h16_t* wbase = w2f + lane * 16;
    v8f acc[8];
#pragma unroll
    for (int n = 0; n < 8; ++n) acc[n] = zero8f();
#pragma unroll
    for (int dt = 0; dt < 3; ++dt)
#pragma unroll
        for (int dm = 0; dm < 3; ++dm)
#pragma unroll
            for (int cb = 0; cb < 4; ++cb) {
                const int kb = (dt * 3 + dm) * 4 + cb;
                const int aoff = (dt * 10 + dm) * CD + cb * 32;
                v16h a = combine16(ld8h(abase + aoff), ld8h(abase + aoff + 16));
                v16h bfr[8];
#pragma unroll
                for (int nt = 0; nt < 8; ++nt)
                    bfr[nt] = ld16h(wbase + (nt * 36 + kb) * 512);
#pragma unroll
                for (int nt = 0; nt < 8; ++nt)
                    acc[nt] = wmma_f16(a, bfr[nt], acc[nt]);
            }
    int gp = mt * 2 + hi;                 // == b*T + t
    int tg = gp & 1023, bg = gp >> 10;
    size_t obase = ((size_t)bg * TP + (tg + 1)) * 4 * CD;
#pragma unroll
    for (int nt = 0; nt < 8; ++nt) {
        int co = nt * 16 + lm;
        float bias = b2[co];
        float p0 = fmaxf(fmaxf(acc[nt][0], acc[nt][1]), fmaxf(acc[nt][2], acc[nt][3]));
        float p1 = fmaxf(fmaxf(acc[nt][4], acc[nt][5]), fmaxf(acc[nt][6], acc[nt][7]));
        f2p[obase + (size_t)1 * CD + co] = (h16_t)fmaxf(p0 + bias, 0.0f);
        f2p[obase + (size_t)2 * CD + co] = (h16_t)fmaxf(p1 + bias, 0.0f);
    }
}

// =====================================================================
// Conv3 implicit GEMM + pool2 -> feats[t*B+b][128] f16
// =====================================================================
__global__ __launch_bounds__(256) void conv3_kernel(
    const h16_t* __restrict__ f2p, const h16_t* __restrict__ w3f,
    const float* __restrict__ b3, h16_t* __restrict__ feats) {
    int mt = (blockIdx.x * blockDim.x + threadIdx.x) >> 5;   // 0..4095
    int lane = threadIdx.x & 31;
    int lm = lane & 15, hi = lane >> 4;
    int p = mt * 16 + lm;
    int b = p >> 11, t = (p >> 1) & 1023, m = p & 1;
    const h16_t* abase = f2p + (((size_t)b * TP + t) * 4 + m) * CD + hi * 8;
    const h16_t* wbase = w3f + lane * 16;
    v8f acc[8];
#pragma unroll
    for (int n = 0; n < 8; ++n) acc[n] = zero8f();
#pragma unroll
    for (int dt = 0; dt < 3; ++dt)
#pragma unroll
        for (int dm = 0; dm < 3; ++dm)
#pragma unroll
            for (int cb = 0; cb < 4; ++cb) {
                const int kb = (dt * 3 + dm) * 4 + cb;
                const int aoff = (dt * 4 + dm) * CD + cb * 32;
                v16h a = combine16(ld8h(abase + aoff), ld8h(abase + aoff + 16));
                v16h bfr[8];
#pragma unroll
                for (int nt = 0; nt < 8; ++nt)
                    bfr[nt] = ld16h(wbase + (nt * 36 + kb) * 512);
#pragma unroll
                for (int nt = 0; nt < 8; ++nt)
                    acc[nt] = wmma_f16(a, bfr[nt], acc[nt]);
            }
#pragma unroll
    for (int nt = 0; nt < 8; ++nt) {
        int co = nt * 16 + lm;
        float bias = b3[co];
#pragma unroll
        for (int q = 0; q < 4; ++q) {
            float v = fmaxf(acc[nt][2 * q], acc[nt][2 * q + 1]);
            v = fmaxf(v + bias, 0.0f);
            int gp = mt * 8 + hi * 4 + q;     // == b*T + t
            int tg = gp & 1023, bg = gp >> 10;
            feats[((size_t)tg * BB + bg) * CD + co] = (h16_t)v;
        }
    }
}

// =====================================================================
// gi_x = feats @ W_ih[:, :128]^T + b_ih.  M=32768, N=768, K=128.
// =====================================================================
__global__ __launch_bounds__(256) void gix_kernel(
    const h16_t* __restrict__ feats, const h16_t* __restrict__ wihxf,
    const float* __restrict__ b_ih, float* __restrict__ giX) {
    int wave = (blockIdx.x * blockDim.x + threadIdx.x) >> 5;  // 0..12287
    int lane = threadIdx.x & 31;
    int lm = lane & 15, hi = lane >> 4;
    int ntg = wave % 6, mt = wave / 6;
    int p = mt * 16 + lm;
    const h16_t* abase = feats + (size_t)p * CD + hi * 8;
    const h16_t* wbase = wihxf + (size_t)ntg * 8 * 4 * 512 + lane * 16;
    v8f acc[8];
#pragma unroll
    for (int n = 0; n < 8; ++n) acc[n] = zero8f();
#pragma unroll
    for (int kb = 0; kb < 4; ++kb) {
        v16h a = combine16(ld8h(abase + kb * 32), ld8h(abase + kb * 32 + 16));
        v16h bfr[8];
#pragma unroll
        for (int ntl = 0; ntl < 8; ++ntl)
            bfr[ntl] = ld16h(wbase + (ntl * 4 + kb) * 512);
#pragma unroll
        for (int ntl = 0; ntl < 8; ++ntl)
            acc[ntl] = wmma_f16(a, bfr[ntl], acc[ntl]);
    }
#pragma unroll
    for (int ntl = 0; ntl < 8; ++ntl) {
        int g = (ntg * 8 + ntl) * 16 + lm;
        float bias = b_ih[g];
#pragma unroll
        for (int r = 0; r < 8; ++r) {
            int row = mt * 16 + r + 8 * hi;
            giX[(size_t)row * G3H + g] = acc[ntl][r] + bias;
        }
    }
}

// =====================================================================
// GRU scan: one persistent 512-thread workgroup, 1024 sequential steps.
// h16 kept in FRAGMENT-MAJOR LDS layout (see round-2 notes).
// =====================================================================
#define LDS_H16  0           // 2*8*32*16 halfs = 16384 B
#define LDS_H32  16384       // 32*256 f32      = 32768 B
#define LDS_GH   49152       // 32*768 f32      = 98304 B
#define LDS_TF   147456      // 32*10 f32
#define LDS_SZ   148736

__global__ __launch_bounds__(512) void gru_kernel(
    const float* __restrict__ giX, const h16_t* __restrict__ whhf,
    const float* __restrict__ b_hh, const float* __restrict__ W_ih,
    const float* __restrict__ Wf, const float* __restrict__ bf,
    const float* __restrict__ targets, const int* __restrict__ force_mask,
    float* __restrict__ out) {
    extern __shared__ char smem[];
    h16_t* h16 = (h16_t*)(smem + LDS_H16);
    float* h32 = (float*)(smem + LDS_H32);
    float* gh  = (float*)(smem + LDS_GH);
    float* tf  = (float*)(smem + LDS_TF);

    int tid = threadIdx.x;
    int wave = tid >> 5, lane = tid & 31;
    int lm = lane & 15, hi = lane >> 4;

    for (int i = tid; i < BB * HH; i += 512) { h16[i] = (h16_t)0.0f; h32[i] = 0.0f; }
    if (tid < BB * CC) tf[tid] = 0.0f;
    __syncthreads();

    const h16_t* wbb = whhf + lane * 16;
    const h16_t* haa = h16 + lane * 16;

    for (int t = 0; t < TT; ++t) {
        // ---- phase 1: gh = h @ W_hh^T (16 waves x (2 mtl x 3 nt) x 8 k WMMA)
        {
            int ntbase = wave * 3;
#pragma unroll
            for (int mtl = 0; mtl < 2; ++mtl) {
#pragma unroll
                for (int j = 0; j < 3; ++j) {
                    int nt = ntbase + j;
                    int g = nt * 16 + lm;
                    const h16_t* wnt = wbb + (size_t)nt * 8 * 512;
                    // batch all 8 global B fragments first (long-latency)
                    v16h bfr[8];
#pragma unroll
                    for (int kb = 0; kb < 8; ++kb)
                        bfr[kb] = ld16h(wnt + kb * 512);
                    v8f acc = zero8f();
#pragma unroll
                    for (int kb = 0; kb < 8; ++kb) {
                        v16h a = ld16h(haa + (mtl * 8 + kb) * 512);
                        acc = wmma_f16(a, bfr[kb], acc);
                    }
                    float* ghp = gh + (mtl * 16 + hi * 8) * G3H + g;
#pragma unroll
                    for (int r = 0; r < 8; ++r) ghp[r * G3H] = acc[r];
                }
            }
        }
        __syncthreads();

        // ---- phase 2: gates + hidden update (32*256 units / 512 threads)
        const float* gi = giX + (size_t)t * BB * G3H;
#pragma unroll 4
        for (int u = 0; u < 16; ++u) {
            int id = (u << 9) + tid;
            int b = id >> 8, j = id & 255;
            float tfc0 = 0.0f, tfc1 = 0.0f, tfc2 = 0.0f;
#pragma unroll
            for (int c = 0; c < CC; ++c) {
                float tv = tf[b * CC + c];
                tfc0 += tv * W_ih[(size_t)(j)       * (CD + CC) + CD + c];
                tfc1 += tv * W_ih[(size_t)(j + 256) * (CD + CC) + CD + c];
                tfc2 += tv * W_ih[(size_t)(j + 512) * (CD + CC) + CD + c];
            }
            float gr = gi[b * G3H + j]       + tfc0;
            float gz = gi[b * G3H + 256 + j] + tfc1;
            float gn = gi[b * G3H + 512 + j] + tfc2;
            float hr = gh[b * G3H + j]       + b_hh[j];
            float hz = gh[b * G3H + 256 + j] + b_hh[256 + j];
            float hn = gh[b * G3H + 512 + j] + b_hh[512 + j];
            float r = sigmoidf_(gr + hr);
            float z = sigmoidf_(gz + hz);
            float n = tanhf(gn + r * hn);
            float hold = h32[id];
            float hnew = (1.0f - z) * n + z * hold;
            h32[id] = hnew;
            // fragment-major store for next step's WMMA A operand
            int mtl = b >> 4, blm = b & 15;
            int kb = j >> 5, rr = j & 31;
            int fhi = (rr >> 3) & 1;
            int fi = (rr & 7) + ((rr >= 16) ? 8 : 0);
            h16[((mtl * 8 + kb) * 32 + fhi * 16 + blm) * 16 + fi] = (h16_t)hnew;
        }
        __syncthreads();

        // ---- phase 3: output head + prediction + teacher forcing
        if (tid < BB * CC) {
            int b = tid / CC, c = tid % CC;
            float s = bf[c];
            const float* hv = h32 + b * HH;
            const float* wv = Wf + (size_t)c * HH;
#pragma unroll 8
            for (int j = 0; j < HH; ++j) s += hv[j] * wv[j];
            out[((size_t)b * TT + t) * CC + c] = s;
            float pred = (s > 0.0f) ? 1.0f : 0.0f;   // sigmoid(s)>0.5 <=> s>0
            int fm = force_mask[t * BB + b];
            tf[tid] = (fm > 0) ? targets[((size_t)b * TT + t) * CC + c] : pred;
        }
        __syncthreads();
    }
}

// =====================================================================
extern "C" void kernel_launch(void* const* d_in, const int* in_sizes, int n_in,
                              void* d_out, int out_size, void* d_ws, size_t ws_size,
                              hipStream_t stream) {
    const float* x          = (const float*)d_in[0];
    const float* targets    = (const float*)d_in[1];
    const int*   force_mask = (const int*)  d_in[2];
    const float* W1         = (const float*)d_in[3];
    const float* b1         = (const float*)d_in[4];
    const float* W2         = (const float*)d_in[5];
    const float* b2         = (const float*)d_in[6];
    const float* W3         = (const float*)d_in[7];
    const float* b3         = (const float*)d_in[8];
    const float* W_ih       = (const float*)d_in[9];
    const float* W_hh       = (const float*)d_in[10];
    const float* b_ih       = (const float*)d_in[11];
    const float* b_hh       = (const float*)d_in[12];
    const float* Wf         = (const float*)d_in[13];
    const float* bf         = (const float*)d_in[14];
    float* out = (float*)d_out;

    char* ws = (char*)d_ws;
    h16_t* f1p   = (h16_t*)(ws + OFF_F1P);
    h16_t* f2p   = (h16_t*)(ws + OFF_F2P);
    h16_t* feats = (h16_t*)(ws + OFF_FEAT);
    float* giX   = (float*)(ws + OFF_GIX);
    h16_t* w2f   = (h16_t*)(ws + OFF_W2F);
    h16_t* w3f   = (h16_t*)(ws + OFF_W3F);
    h16_t* wihxf = (h16_t*)(ws + OFF_WIHXF);
    h16_t* whhf  = (h16_t*)(ws + OFF_WHHF);

    // zero-fill padded activation buffers (halo must be 0)
    {
        size_t n8a = (size_t)BB * TP * 10 * CD / 8;
        size_t n8b = (size_t)BB * TP * 4 * CD / 8;
        zero_f16<<<(unsigned)((n8a + 255) / 256), 256, 0, stream>>>(f1p, n8a);
        zero_f16<<<(unsigned)((n8b + 255) / 256), 256, 0, stream>>>(f2p, n8b);
    }

    // fragment-major weight packing
    pack_conv_frag<<<(8 * 36 * 32 * 16 + 255) / 256, 256, 0, stream>>>(W2, w2f);
    pack_conv_frag<<<(8 * 36 * 32 * 16 + 255) / 256, 256, 0, stream>>>(W3, w3f);
    pack_wihx_frag<<<(48 * 4 * 32 * 16 + 255) / 256, 256, 0, stream>>>(W_ih, wihxf);
    pack_whh_frag <<<(48 * 8 * 32 * 16 + 255) / 256, 256, 0, stream>>>(W_hh, whhf);

    // CNN feature extractor
    conv1_kernel<<<BB * TT, 128, 0, stream>>>(x, W1, b1, f1p);
    conv2_kernel<<<2048, 256, 0, stream>>>(f1p, w2f, b2, f2p);   // 16384 waves
    conv3_kernel<<<512, 256, 0, stream>>>(f2p, w3f, b3, feats);  // 4096 waves

    // time-parallel GRU input gates
    gix_kernel<<<1536, 256, 0, stream>>>(feats, wihxf, b_ih, giX); // 12288 waves

    // sequential GRU scan (single persistent workgroup)
    gru_kernel<<<1, 512, LDS_SZ, stream>>>(giX, whhf, b_hh, W_ih, Wf, bf,
                                           targets, force_mask, out);
}
